// DeepFM_69286412419117
// MI455X (gfx1250) — matrix operand order repaired and checked by reading, not compiled, not run on hardware
//
#include <hip/hip_runtime.h>

typedef __attribute__((ext_vector_type(16))) _Float16 v16h;
typedef __attribute__((ext_vector_type(8)))  float    v8f;

#define FIELD 39
#define VOCAB 100000
#define EMB   32
#define H1N   32
#define H2N   32
#define BATCH 16384
#define CH    13          // fields staged per LDS chunk (39 = 3 * 13)
#define WAVES 8           // waves per block (wave32)
#define TPB   (WAVES * 32)

__global__ __launch_bounds__(TPB) void deepfm_fused_kernel(
    const int*   __restrict__ Xi,   const float* __restrict__ Xv,
    const float* __restrict__ W1,   const float* __restrict__ W2,
    const float* __restrict__ l1w,  const float* __restrict__ l1b,
    const float* __restrict__ l2w,  const float* __restrict__ l2b,
    const float* __restrict__ bias, float* __restrict__ out)
{
    // lin1_w chunk pre-packed into WMMA B-fragment layout (f16):
    // index = ((fi*2 + ntile)*32 + lane)*16 + j  ->  lin1_w[k][n],
    //   k = (f0+fi)*32 + (lane>=16 ? 8 : 0) + (j&7) + (j>=8 ? 16 : 0),  n = ntile*16 + (lane&15)
    __shared__ __align__(32) _Float16 lds_b [CH * 2 * 32 * 16];   // 26 KB
    __shared__ __align__(32) _Float16 lds_b2[2 * 32 * 16];        //  2 KB
    __shared__ __align__(16) _Float16 lds_h1[WAVES][16 * 32];     //  8 KB
    __shared__ float                  lds_rows[WAVES][16];        // .5 KB

    const int tid  = threadIdx.x;
    const int wave = tid >> 5;
    const int l    = tid & 31;
    const int m    = l & 15;         // batch row within tile
    const int hi   = l >> 4;         // which half of the K/e range this lane owns
    const int r0   = (blockIdx.x * WAVES + wave) * 16;
    const int r    = r0 + m;

    // Stage layer-2 weights (B-fragment layout, f16) once.
    for (int i = tid; i < 2 * 32 * 16; i += TPB) {
        int t = (i >> 9) & 1, ll = (i >> 4) & 31, j = i & 15;
        int k = ((ll >> 4) << 3) + (j & 7) + ((j >> 3) << 4);
        int n = t * 16 + (ll & 15);
        lds_b2[i] = (_Float16)l2w[k * H2N + n];
    }

    v8f c0 = {}, c1 = {};                 // layer-1 f32 accumulators (N 0..15 / 16..31)
    float s_acc[16], sq_acc[16];          // FM: sum(v) and sum(v^2) for this lane's 16 e's
    #pragma unroll
    for (int i = 0; i < 16; ++i) { s_acc[i] = 0.f; sq_acc[i] = 0.f; }
    float facc = 0.f;                     // first-order term (row m)

    const int g0f4 = hi ? 2 : 0;          // float4 group base inside a 32-float embedding

    for (int c = 0; c < FIELD / CH; ++c) {
        const int f0 = c * CH;
        __syncthreads();
        // Stage lin1_w chunk into B-fragment f16 layout.
        for (int i = tid; i < CH * 1024; i += TPB) {
            int fi = i >> 10, t = (i >> 9) & 1, ll = (i >> 4) & 31, j = i & 15;
            int k = (f0 + fi) * 32 + ((ll >> 4) << 3) + (j & 7) + ((j >> 3) << 4);
            int n = t * 16 + (ll & 15);
            lds_b[i] = (_Float16)l1w[k * H1N + n];
        }
        __syncthreads();

        for (int fi = 0; fi < CH; ++fi) {
            const int   f   = f0 + fi;
            const int   idx = Xi[(size_t)r * FIELD + f];
            const float xv  = Xv[(size_t)r * FIELD + f];
            facc += W1[(size_t)f * VOCAB + idx] * xv;

            // Gather this lane's half of the embedding row, in A-fragment order:
            // lane<16 : e in {0..7, 16..23}; lane>=16 : e in {8..15, 24..31}
            const float4* w2p = (const float4*)(W2 + ((size_t)f * VOCAB + idx) * EMB);
            float4 q0 = w2p[g0f4],     q1 = w2p[g0f4 + 1];
            float4 q2 = w2p[g0f4 + 4], q3 = w2p[g0f4 + 5];
            float e[16] = { q0.x,q0.y,q0.z,q0.w, q1.x,q1.y,q1.z,q1.w,
                            q2.x,q2.y,q2.z,q2.w, q3.x,q3.y,q3.z,q3.w };
            v16h a;
            #pragma unroll
            for (int i = 0; i < 16; ++i) {
                float v = e[i] * xv;
                s_acc[i]  += v;
                sq_acc[i] += v * v;
                a[i] = (_Float16)v;
            }
            const v16h b0 = *(const v16h*)(lds_b + ((fi * 2 + 0) * 32 + l) * 16);
            const v16h b1 = *(const v16h*)(lds_b + ((fi * 2 + 1) * 32 + l) * 16);
            c0 = __builtin_amdgcn_wmma_f32_16x16x32_f16(false, a, false, b0, (short)0, c0, false, false);
            c1 = __builtin_amdgcn_wmma_f32_16x16x32_f16(false, a, false, b1, (short)0, c1, false, false);
        }
    }

    // FM second-order term: 0.5 * (sum_e S^2 - sum_e SQ), e split across lane pair (l, l^16).
    float fm = 0.f;
    #pragma unroll
    for (int i = 0; i < 16; ++i) fm += s_acc[i] * s_acc[i] - sq_acc[i];
    fm += __shfl_xor(fm, 16, 32);
    const float fs = facc + 0.5f * fm;

    // Layer-1 bias + relu; stage activations [m][n] (f16) for layer-2 A fragment.
    const int   nl  = l & 15;
    const float b1a = l1b[nl], b1b = l1b[nl + 16];
    const int   mb  = hi * 8;
    #pragma unroll
    for (int v = 0; v < 8; ++v) {
        float h0 = c0[v] + b1a; h0 = h0 > 0.f ? h0 : 0.f;
        float h1 = c1[v] + b1b; h1 = h1 > 0.f ? h1 : 0.f;
        lds_h1[wave][(mb + v) * 32 + nl]      = (_Float16)h0;
        lds_h1[wave][(mb + v) * 32 + nl + 16] = (_Float16)h1;
    }
    __syncthreads();

    // Layer-2: rebuild A fragment from LDS, two WMMAs.
    const _Float16* h1p = lds_h1[wave] + m * 32;
    const int e0 = hi * 8;
    v16h a2;
    #pragma unroll
    for (int i = 0; i < 8; ++i) {
        a2[i]     = h1p[e0 + i];
        a2[i + 8] = h1p[e0 + 16 + i];
    }
    const v16h b20 = *(const v16h*)(lds_b2 + (0 * 32 + l) * 16);
    const v16h b21 = *(const v16h*)(lds_b2 + (1 * 32 + l) * 16);
    v8f z = {};
    v8f d0 = __builtin_amdgcn_wmma_f32_16x16x32_f16(false, a2, false, b20, (short)0, z, false, false);
    v8f d1 = __builtin_amdgcn_wmma_f32_16x16x32_f16(false, a2, false, b21, (short)0, z, false, false);

    // Layer-2 bias + relu, then row-sum over N via butterfly within each 16-lane half.
    const float b2a = l2b[nl], b2b = l2b[nl + 16];
    float rowsum[8];
    #pragma unroll
    for (int v = 0; v < 8; ++v) {
        float x0 = d0[v] + b2a; x0 = x0 > 0.f ? x0 : 0.f;
        float x1 = d1[v] + b2b; x1 = x1 > 0.f ? x1 : 0.f;
        float rv = x0 + x1;
        rv += __shfl_xor(rv, 1, 32);
        rv += __shfl_xor(rv, 2, 32);
        rv += __shfl_xor(rv, 4, 32);
        rv += __shfl_xor(rv, 8, 32);
        rowsum[v] = rv;   // rows mb..mb+7, valid on every lane in this half
    }
    if (nl == 0) {
        #pragma unroll
        for (int v = 0; v < 8; ++v) lds_rows[wave][mb + v] = rowsum[v];
    }
    __syncthreads();

    if (l < 16) {
        out[r0 + l] = fs + lds_rows[wave][l] + bias[0];
    }
}

extern "C" void kernel_launch(void* const* d_in, const int* in_sizes, int n_in,
                              void* d_out, int out_size, void* d_ws, size_t ws_size,
                              hipStream_t stream) {
    const int*   Xi  = (const int*)d_in[0];
    const float* Xv  = (const float*)d_in[1];
    const float* W1  = (const float*)d_in[2];
    const float* W2  = (const float*)d_in[3];
    const float* l1w = (const float*)d_in[4];
    const float* l1b = (const float*)d_in[5];
    const float* l2w = (const float*)d_in[6];
    const float* l2b = (const float*)d_in[7];
    const float* bs  = (const float*)d_in[8];
    float* out = (float*)d_out;

    dim3 grid(BATCH / (16 * WAVES));   // 128 blocks x 256 threads (8 waves of 16 rows)
    deepfm_fused_kernel<<<grid, TPB, 0, stream>>>(Xi, Xv, W1, W2, l1w, l1b, l2w, l2b, bs, out);
}
